// EffAttention_28862180229728
// MI455X (gfx1250) — matrix operand order repaired and checked by reading, not compiled, not run on hardware
//
#include <hip/hip_runtime.h>
#include <hip/hip_bf16.h>

typedef __attribute__((ext_vector_type(16))) __bf16 v16bf;
typedef __attribute__((ext_vector_type(8)))  __bf16 v8bf;
typedef __attribute__((ext_vector_type(4)))  __bf16 v4bf;
typedef __attribute__((ext_vector_type(8)))  float  v8f;

// Assemble a 16-element fragment from two contiguous 16-byte LDS runs
// (each lowers to ds_load_b128).
static __device__ __forceinline__ v16bf frag2(const __bf16* lo, const __bf16* hi) {
    v8bf l = *(const v8bf*)lo;
    v8bf h = *(const v8bf*)hi;
    v16bf f;
#pragma unroll
    for (int t = 0; t < 8; ++t) { f[t] = l[t]; f[t + 8] = h[t]; }
    return f;
}

static __device__ __forceinline__ v4bf cvt4(float4 v) {
    v4bf p;
    p[0] = (__bf16)v.x; p[1] = (__bf16)v.y; p[2] = (__bf16)v.z; p[3] = (__bf16)v.w;
    return p;
}

// ---------------------------------------------------------------------------
// Tiled GEMM: C[M,N] = A x B[K,N] (+bias). bf16 WMMA, f32 accumulate.
// MODE 0: A is [M,K] row-major, lda = K            (vector loads along k)
// MODE 1: A is [K,M] row-major, lda = M-stride     (vector loads along m)
// Tile 128x128, k-step 32; 256 thr = 8 waves; each wave owns a 32x64 patch
// = 2 A-frags x 4 B-frags = 8 WMMAs per k-step. Register double-buffering
// of the next k-step's global float4 loads.
// LDS: sA[m][k] (A-frag = 2 runs of 8), sB[n][k] (B-frag = 2 runs of 8),
// rows padded to 40 bf16 (80 B) so every fragment run is 16-byte aligned.
// All launch shapes divide tiles exactly -> no bounds checks.
// ---------------------------------------------------------------------------
template<int MODE>
__global__ __launch_bounds__(256) void gemm_bf16_wmma(
    const float* __restrict__ A, const float* __restrict__ Bw,
    const float* __restrict__ bias, float* __restrict__ C,
    int M, int N, int K, long lda, long abatch, long cbatch)
{
    __shared__ __attribute__((aligned(16))) __bf16 sA[128][40];
    __shared__ __attribute__((aligned(16))) __bf16 sB[128][40];

    const float* Ab = A + (long)blockIdx.z * abatch;
    float*       Cb = C + (long)blockIdx.z * cbatch;
    const int m0 = blockIdx.y * 128, n0 = blockIdx.x * 128;
    const int tid = threadIdx.x, lane = tid & 31, wave = tid >> 5;
    const int wm2 = (wave & 3) * 2;        // two m-tiles: wm2, wm2+1
    const int wn4 = (wave >> 2) * 4;       // four n-tiles: wn4..wn4+3
    const int mrow = lane & 15, hi = lane >> 4;

    float4 aReg[4], bReg[4];

    auto loadG = [&](int k0) {
#pragma unroll
        for (int s = 0; s < 4; ++s) {
            int i = tid + s * 256;               // 1024 float4 per tile
            if (MODE == 0) {
                int r = i >> 3, c4 = (i & 7) * 4;
                aReg[s] = *(const float4*)(Ab + (long)(m0 + r) * lda + k0 + c4);
            } else {
                int kk = i >> 5, m4 = (i & 31) * 4;
                aReg[s] = *(const float4*)(Ab + (long)(k0 + kk) * lda + m0 + m4);
            }
            int kk = i >> 5, n4 = (i & 31) * 4;
            bReg[s] = *(const float4*)(Bw + (long)(k0 + kk) * N + n0 + n4);
        }
    };
    auto storeL = [&]() {
#pragma unroll
        for (int s = 0; s < 4; ++s) {
            int i = tid + s * 256;
            if (MODE == 0) {
                int r = i >> 3, c4 = (i & 7) * 4;
                *(v4bf*)&sA[r][c4] = cvt4(aReg[s]);
            } else {
                int kk = i >> 5, m4 = (i & 31) * 4;
                sA[m4 + 0][kk] = (__bf16)aReg[s].x;
                sA[m4 + 1][kk] = (__bf16)aReg[s].y;
                sA[m4 + 2][kk] = (__bf16)aReg[s].z;
                sA[m4 + 3][kk] = (__bf16)aReg[s].w;
            }
            int kk = i >> 5, n4 = (i & 31) * 4;
            sB[n4 + 0][kk] = (__bf16)bReg[s].x;  // transpose B -> [n][k]
            sB[n4 + 1][kk] = (__bf16)bReg[s].y;
            sB[n4 + 2][kk] = (__bf16)bReg[s].z;
            sB[n4 + 3][kk] = (__bf16)bReg[s].w;
        }
    };

    v8f acc[2][4] = {};
    loadG(0);
    for (int k0 = 0; k0 < K; k0 += 32) {
        __syncthreads();                         // previous tile fully consumed
        storeL();
        __syncthreads();
        if (k0 + 32 < K) loadG(k0 + 32);         // overlap with WMMA

        v16bf af[2], bf[4];
#pragma unroll
        for (int im = 0; im < 2; ++im) {
            const int ar = (wm2 + im) * 16 + mrow;
            af[im] = frag2(&sA[ar][8 * hi], &sA[ar][16 + 8 * hi]);
        }
#pragma unroll
        for (int jn = 0; jn < 4; ++jn) {
            const int br = (wn4 + jn) * 16 + mrow;
            bf[jn] = frag2(&sB[br][8 * hi], &sB[br][16 + 8 * hi]);
        }
#pragma unroll
        for (int im = 0; im < 2; ++im)
#pragma unroll
            for (int jn = 0; jn < 4; ++jn)
                acc[im][jn] = __builtin_amdgcn_wmma_f32_16x16x32_bf16(
                    false, af[im], false, bf[jn], (short)0, acc[im][jn],
                    false, false);
    }

#pragma unroll
    for (int jn = 0; jn < 4; ++jn) {
        const int gn = n0 + (wn4 + jn) * 16 + mrow;
        const float bv = bias ? bias[gn] : 0.f;
#pragma unroll
        for (int im = 0; im < 2; ++im) {
            const int gmb = m0 + (wm2 + im) * 16 + 8 * hi;
#pragma unroll
            for (int v = 0; v < 8; ++v)          // C-layout: m = v + 8*hi
                Cb[(long)(gmb + v) * N + gn] = acc[im][jn][v] + bv;
        }
    }
}

// ---------------------------------------------------------------------------
// qkv_mat [B,4096,768] -> cat image layout [B,3072,4096], channels 0..767
// ---------------------------------------------------------------------------
__global__ __launch_bounds__(256) void transpose_qkv(
    const float* __restrict__ qkv, float* __restrict__ cat)
{
    __shared__ float tile[32][33];
    const int p0 = blockIdx.x * 32, c0 = blockIdx.y * 32, b = blockIdx.z;
    const int tx = threadIdx.x, ty = threadIdx.y;
    const float* src = qkv + (long)b * 4096 * 768;
    float*       dst = cat + (long)b * 3072 * 4096;
    for (int j = ty; j < 32; j += 8)
        tile[j][tx] = src[(long)(p0 + j) * 768 + c0 + tx];
    __syncthreads();
    for (int j = ty; j < 32; j += 8)
        dst[(long)(c0 + j) * 4096 + p0 + tx] = tile[tx][j];
}

// ---------------------------------------------------------------------------
// Depthwise KSxKS SAME conv over 64x64 image; input = cat channels 0..767.
// ---------------------------------------------------------------------------
template<int KS>
__global__ __launch_bounds__(256) void dwconv(
    const float* __restrict__ cat, const float* __restrict__ w,
    float* __restrict__ out)
{
    long idx = (long)blockIdx.x * 256 + threadIdx.x;   // over 2*768*4096
    int p = (int)(idx & 4095);
    int c = (int)((idx >> 12) % 768);
    int b = (int)(idx / (768L * 4096));
    int x = p & 63, y = p >> 6;
    const int R = KS >> 1;
    const float* ib = cat + ((long)b * 3072 + c) * 4096;
    const float* wc = w + (long)c * (KS * KS);
    float acc = 0.f;
#pragma unroll
    for (int dy = -R; dy <= R; ++dy) {
        int yy = y + dy;
        if (yy < 0 || yy > 63) continue;
#pragma unroll
        for (int dx = -R; dx <= R; ++dx) {
            int xx = x + dx;
            if (xx < 0 || xx > 63) continue;
            acc += ib[yy * 64 + xx] * wc[(dy + R) * KS + (dx + R)];
        }
    }
    out[((long)b * 768 + c) * 4096 + p] = acc;
}

// ---------------------------------------------------------------------------
// Grouped 1x1 conv (24 groups of 32ch): per (b,g) a [4096x32]@[32x32] GEMM,
// K=32 == one bf16 WMMA per wave tile. LDS: sA[pixel][ic], sB[oc][ic].
// ---------------------------------------------------------------------------
__global__ __launch_bounds__(256) void pwconv_wmma(
    const float* __restrict__ in, const float* __restrict__ pw,
    float* __restrict__ cat, int branch)
{
    __shared__ __attribute__((aligned(16))) __bf16 sA[64][40];  // [pix][ic]
    __shared__ __attribute__((aligned(16))) __bf16 sB[32][40];  // [oc][ic]
    const int b = blockIdx.z, g = blockIdx.y, p0 = blockIdx.x * 64;
    const int tid = threadIdx.x, lane = tid & 31, wave = tid >> 5;
    const float* ib = in + ((long)b * 768 + g * 32) * 4096;

#pragma unroll
    for (int s = 0; s < 2; ++s) {                 // 512 float4 of A
        int i = tid + s * 256;
        int ic = i >> 4, p4 = (i & 15) * 4;
        float4 v = *(const float4*)(ib + (long)ic * 4096 + p0 + p4);
        sA[p4 + 0][ic] = (__bf16)v.x;
        sA[p4 + 1][ic] = (__bf16)v.y;
        sA[p4 + 2][ic] = (__bf16)v.z;
        sA[p4 + 3][ic] = (__bf16)v.w;
    }
    {                                              // 256 float4 of W
        int o = tid >> 3, ic4 = (tid & 7) * 4;
        float4 v = *(const float4*)(pw + (long)(g * 32 + o) * 32 + ic4);
        *(v4bf*)&sB[o][ic4] = cvt4(v);
    }
    __syncthreads();

    const int mt = wave >> 1, nt = wave & 1;
    const int mrow = lane & 15, hi = lane >> 4;
    const int ar = mt * 16 + mrow, br = nt * 16 + mrow;
    v16bf af = frag2(&sA[ar][8 * hi], &sA[ar][16 + 8 * hi]);
    v16bf bf = frag2(&sB[br][8 * hi], &sB[br][16 + 8 * hi]);
    v8f acc = {};
    acc = __builtin_amdgcn_wmma_f32_16x16x32_bf16(false, af, false, bf,
                                                  (short)0, acc, false, false);
    long chbase = ((long)b * 3072 + branch * 768 + g * 32 + br) * 4096;
#pragma unroll
    for (int v = 0; v < 8; ++v)
        cat[chbase + p0 + mt * 16 + 8 * hi + v] = acc[v];
}

// ---------------------------------------------------------------------------
// Flash attention within 2048-token chunks, hd=32 (== bf16 WMMA K).
// Grid (16 q-tiles of 128 rows, 8 heads, b*2+chunk). 256 thr = 8 waves; each
// wave owns a 16-row Q fragment. Per 32-key step: 2 QK^T WMMAs -> online
// softmax (shuffle row-reduce in the 16-lane half owning each row) -> P
// restaged [m][k] via per-wave LDS (s_wait_dscnt) -> 2 PV WMMAs.
// K/V global loads double-buffered in registers.
// ---------------------------------------------------------------------------
__global__ __launch_bounds__(256) void flash_attn(
    const float* __restrict__ msq, float* __restrict__ out)
{
    const int h = blockIdx.y;
    const int b = blockIdx.z >> 1, chunk = blockIdx.z & 1;
    const int q0 = chunk * 2048 + blockIdx.x * 128;
    const int tid = threadIdx.x, lane = tid & 31, wave = tid >> 5;
    const int mrow = lane & 15, hi = lane >> 4;
    const float scale = 0.17677669529663687f;     // 32^-0.5

    __shared__ __attribute__((aligned(16))) __bf16 sQ[128][40];     // [m][d]
    __shared__ __attribute__((aligned(16))) __bf16 sK[32][40];      // [key][d]
    __shared__ __attribute__((aligned(16))) __bf16 sVt[32][40];     // [d][key]
    __shared__ __attribute__((aligned(16))) __bf16 sP[8][16][40];   // [wave][m][k]

    const float* base = msq + (long)b * 4096 * 768;

#pragma unroll
    for (int s = 0; s < 4; ++s) {                 // Q tile: 1024 float4
        int i = tid + s * 256;
        int r = i >> 3, c4 = (i & 7) * 4;
        float4 v = *(const float4*)(base + (long)(q0 + r) * 768 + h * 32 + c4);
        v4bf p;
        p[0] = (__bf16)(v.x * scale); p[1] = (__bf16)(v.y * scale);
        p[2] = (__bf16)(v.z * scale); p[3] = (__bf16)(v.w * scale);
        *(v4bf*)&sQ[r][c4] = p;
    }
    __syncthreads();

    const int qr = wave * 16 + mrow;
    v16bf qf = frag2(&sQ[qr][8 * hi], &sQ[qr][16 + 8 * hi]);

    v8f o0 = {}, o1 = {};
    float mrun[8], lrun[8];
#pragma unroll
    for (int v = 0; v < 8; ++v) { mrun[v] = -3.0e38f; lrun[v] = 0.f; }

    const int kbase = chunk * 2048;
    float4 kReg, vReg;
    const int kvr = tid >> 3, kvc = (tid & 7) * 4;   // 256 float4 = 1/thread
    auto loadKV = [&](int kt) {
        long row = (long)(kbase + kt + kvr) * 768 + h * 32 + kvc;
        kReg = *(const float4*)(base + row + 256);
        vReg = *(const float4*)(base + row + 512);
    };
    loadKV(0);

    for (int kt = 0; kt < 2048; kt += 32) {
        __syncthreads();
        *(v4bf*)&sK[kvr][kvc] = cvt4(kReg);
        sVt[kvc + 0][kvr] = (__bf16)vReg.x;        // transpose V -> [d][key]
        sVt[kvc + 1][kvr] = (__bf16)vReg.y;
        sVt[kvc + 2][kvr] = (__bf16)vReg.z;
        sVt[kvc + 3][kvr] = (__bf16)vReg.w;
        __syncthreads();
        if (kt + 32 < 2048) loadKV(kt + 32);

        v16bf kf0 = frag2(&sK[mrow][8 * hi],      &sK[mrow][16 + 8 * hi]);
        v16bf kf1 = frag2(&sK[16 + mrow][8 * hi], &sK[16 + mrow][16 + 8 * hi]);
        v16bf vf0 = frag2(&sVt[mrow][8 * hi],      &sVt[mrow][16 + 8 * hi]);
        v16bf vf1 = frag2(&sVt[16 + mrow][8 * hi], &sVt[16 + mrow][16 + 8 * hi]);

        v8f s0 = {}, s1 = {};
        s0 = __builtin_amdgcn_wmma_f32_16x16x32_bf16(false, qf, false, kf0,
                                                     (short)0, s0, false, false);
        s1 = __builtin_amdgcn_wmma_f32_16x16x32_bf16(false, qf, false, kf1,
                                                     (short)0, s1, false, false);
#pragma unroll
        for (int v = 0; v < 8; ++v) {
            float mx = fmaxf(s0[v], s1[v]);
            mx = fmaxf(mx, __shfl_xor(mx, 1));
            mx = fmaxf(mx, __shfl_xor(mx, 2));
            mx = fmaxf(mx, __shfl_xor(mx, 4));
            mx = fmaxf(mx, __shfl_xor(mx, 8));
            float mnew = fmaxf(mrun[v], mx);
            float p0 = __expf(s0[v] - mnew);
            float p1 = __expf(s1[v] - mnew);
            float rs = p0 + p1;
            rs += __shfl_xor(rs, 1);
            rs += __shfl_xor(rs, 2);
            rs += __shfl_xor(rs, 4);
            rs += __shfl_xor(rs, 8);
            float alpha = __expf(mrun[v] - mnew);
            lrun[v] = lrun[v] * alpha + rs;
            mrun[v] = mnew;
            o0[v] *= alpha;
            o1[v] *= alpha;
            int m = v + 8 * hi;                    // C-layout row
            sP[wave][m][mrow]      = (__bf16)p0;   // stage P as [m][k]
            sP[wave][m][16 + mrow] = (__bf16)p1;
        }
        asm volatile("s_wait_dscnt 0x0" ::: "memory");   // wave-local LDS RAW
        v16bf pf = frag2(&sP[wave][mrow][8 * hi], &sP[wave][mrow][16 + 8 * hi]);
        o0 = __builtin_amdgcn_wmma_f32_16x16x32_bf16(false, pf, false, vf0,
                                                     (short)0, o0, false, false);
        o1 = __builtin_amdgcn_wmma_f32_16x16x32_bf16(false, pf, false, vf1,
                                                     (short)0, o1, false, false);
    }

#pragma unroll
    for (int v = 0; v < 8; ++v) {
        float inv = 1.0f / lrun[v];
        int token = q0 + wave * 16 + v + 8 * hi;
        long o = ((long)b * 4096 + token) * 256 + h * 32;
        out[o + mrow]      = o0[v] * inv;
        out[o + 16 + mrow] = o1[v] * inv;
    }
}

// ---------------------------------------------------------------------------
extern "C" void kernel_launch(void* const* d_in, const int* in_sizes, int n_in,
                              void* d_out, int out_size, void* d_ws, size_t ws_size,
                              hipStream_t stream)
{
    const float* x         = (const float*)d_in[0];
    const float* w_reduce  = (const float*)d_in[1];
    const float* w_qkv     = (const float*)d_in[2];
    const float* dw0 = (const float*)d_in[3];
    const float* pw0 = (const float*)d_in[4];
    const float* dw1 = (const float*)d_in[5];
    const float* pw1 = (const float*)d_in[6];
    const float* dw2 = (const float*)d_in[7];
    const float* pw2 = (const float*)d_in[8];
    const float* w_reduce2 = (const float*)d_in[9];
    const float* w_proj    = (const float*)d_in[10];
    const float* b_proj    = (const float*)d_in[11];
    float* out = (float*)d_out;

    float* ws     = (float*)d_ws;
    float* xr     = ws;                       // [8192,256]
    float* qkvm   = xr    + 8192L * 256;      // [8192,768]
    float* cat    = qkvm  + 8192L * 768;      // [2,3072,4096]
    float* dwtmp  = cat   + 2L * 3072 * 4096; // [2,768,4096] (reused per branch)
    float* msq    = dwtmp + 2L * 768 * 4096;  // [8192,768]
    float* attn_o = msq   + 8192L * 768;      // [8192,256]

    // 1) xr = x @ w_reduce                 [8192,512]x[512,256]
    gemm_bf16_wmma<0><<<dim3(2, 64, 1), 256, 0, stream>>>(
        x, w_reduce, nullptr, xr, 8192, 256, 512, 512, 0, 0);
    // 2) qkv = xr @ w_qkv                  [8192,256]x[256,768]
    gemm_bf16_wmma<0><<<dim3(6, 64, 1), 256, 0, stream>>>(
        xr, w_qkv, nullptr, qkvm, 8192, 768, 256, 256, 0, 0);
    // 3) branch 0 of cat = qkv in NCHW image layout
    transpose_qkv<<<dim3(128, 24, 2), dim3(32, 8), 0, stream>>>(qkvm, cat);
    // 4) three dw->pw branches into cat channels [768..3072)
    dwconv<3><<<24576, 256, 0, stream>>>(cat, dw0, dwtmp);
    pwconv_wmma<<<dim3(64, 24, 2), 256, 0, stream>>>(dwtmp, pw0, cat, 1);
    dwconv<5><<<24576, 256, 0, stream>>>(cat, dw1, dwtmp);
    pwconv_wmma<<<dim3(64, 24, 2), 256, 0, stream>>>(dwtmp, pw1, cat, 2);
    dwconv<7><<<24576, 256, 0, stream>>>(cat, dw2, dwtmp);
    pwconv_wmma<<<dim3(64, 24, 2), 256, 0, stream>>>(dwtmp, pw2, cat, 3);
    // 5) msq = cat^T @ w_reduce2 (A stored [K=3072ch][M=4096px] per batch)
    gemm_bf16_wmma<1><<<dim3(6, 32, 2), 256, 0, stream>>>(
        cat, w_reduce2, nullptr, msq, 4096, 768, 3072,
        4096, 3072L * 4096, 4096L * 768);
    // 6) chunked flash attention
    flash_attn<<<dim3(16, 8, 4), 256, 0, stream>>>(msq, attn_o);
    // 7) out = attn_o @ w_proj + b_proj
    gemm_bf16_wmma<0><<<dim3(4, 64, 1), 256, 0, stream>>>(
        attn_o, w_proj, b_proj, out, 8192, 512, 256, 256, 0, 0);
}
